// Set2Set_8967891714154
// MI455X (gfx1250) — matrix operand order repaired and checked by reading, not compiled, not run on hardware
//
#include <hip/hip_runtime.h>

#define NN 131072      // nodes
#define D  512         // channels
#define NB 512         // graphs
#define T_STEPS 6
#define KD 1024        // folded GEMM K  ([h | r])
#define GD 2048        // gate dim (4*D)

typedef __bf16 v16bf __attribute__((ext_vector_type(16)));
typedef __bf16 v8bf  __attribute__((ext_vector_type(8)));
typedef float  v8f   __attribute__((ext_vector_type(8)));
typedef float  v4f   __attribute__((ext_vector_type(4)));
typedef unsigned short u16;

__device__ __forceinline__ u16 f2bf(float f) {
    union { float f; unsigned u; } v; v.f = f;
    unsigned r = 0x7FFFu + ((v.u >> 16) & 1u);
    return (u16)((v.u + r) >> 16);
}

__device__ __forceinline__ float fast_sigmoid(float x) {
    return 1.f / (1.f + __expf(-x));          // v_exp_f32 path, no branches
}
__device__ __forceinline__ float fast_tanh(float x) {
    return 2.f / (1.f + __expf(-2.f * x)) - 1.f;  // stable at +/-inf
}

// ---------------- prep: fold W_ih/W_hh into bf16 [2048 x 1024], bias ----------------
__global__ __launch_bounds__(256) void prep_weights(
    const float* __restrict__ Wih, const float* __restrict__ Whh,
    const float* __restrict__ bih, const float* __restrict__ bhh,
    u16* __restrict__ Wbf, float* __restrict__ bias)
{
    int idx = blockIdx.x * 256 + threadIdx.x;
    int stride = gridDim.x * 256;
    for (int i = idx; i < GD * KD; i += stride) {
        int j = i >> 10, k = i & 1023;
        float w = Wih[j * 1024 + k];           // W_ih is [2048 x 1024]
        if (k < D) w += Whh[j * D + k];        // fold W_hh into q-columns (q == h)
        Wbf[i] = f2bf(w);
    }
    if (idx < GD) bias[idx] = bih[idx] + bhh[idx];
}

// ---------------- segment offsets by binary search over sorted batch ----------------
__global__ void build_seg(const int* __restrict__ batch, int* __restrict__ seg) {
    int b = blockIdx.x * blockDim.x + threadIdx.x;
    if (b > NB) return;
    int lo = 0, hi = NN;
    while (lo < hi) { int mid = (lo + hi) >> 1; if (batch[mid] < b) lo = mid + 1; else hi = mid; }
    seg[b] = lo;   // seg[b] = lower_bound(batch, b); seg[NB] = NN
}

__global__ __launch_bounds__(256) void init_state(float* __restrict__ c, u16* __restrict__ Abf0) {
    int idx = blockIdx.x * 256 + threadIdx.x;
    int stride = gridDim.x * 256;
    for (int i = idx; i < NB * D; i += stride) c[i] = 0.f;
    for (int i = idx; i < NB * KD; i += stride) Abf0[i] = 0;
}

// ---------------- LSTM cell: WMMA GEMM [512 x 1024] x [1024 x 2048] + fused gate epilogue
// Each wave owns one 16x16 (b, d) block and accumulates its i/f/g/o gate tiles
// (columns d0, d0+512, d0+1024, d0+1536) so the cell update fuses in-register.
__global__ __launch_bounds__(256) void lstm_kernel(
    const u16*  __restrict__ Abf_in,   // [512 x 1024] bf16  = [h | r] of prev step
    const u16*  __restrict__ Wbf,      // [2048 x 1024] bf16
    const float* __restrict__ bias,    // [2048]
    float* __restrict__ c,             // [512 x 512] in/out
    float* __restrict__ h,             // [512 x 512] out (q for attention)
    u16*  __restrict__ Abf_out)        // write h_bf into cols [0,512)
{
    const int lane  = threadIdx.x & 31;
    const int wave  = threadIdx.x >> 5;
    const int d_grp = blockIdx.x & 3;
    const int b_blk = blockIdx.x >> 2;
    const int b0 = b_blk * 16;
    const int d0 = d_grp * 128 + wave * 16;

    const int rowA = b0 + (lane & 15);     // A frag: lane -> matrix row M
    const int hs   = lane >> 4;            // lane half selects K sub-ranges
    const int colB = lane & 15;            // B frag: lane -> matrix column N

    const u16* __restrict__ Arow = Abf_in + (size_t)rowA * KD;
    const u16* __restrict__ Wi = Wbf + (size_t)(d0 +        colB) * KD + hs * 16;
    const u16* __restrict__ Wf = Wbf + (size_t)(d0 +  512 + colB) * KD + hs * 16;
    const u16* __restrict__ Wg = Wbf + (size_t)(d0 + 1024 + colB) * KD + hs * 16;
    const u16* __restrict__ Wo = Wbf + (size_t)(d0 + 1536 + colB) * KD + hs * 16;

    v8f acc_i = {}, acc_f = {}, acc_g = {}, acc_o = {};

    #pragma unroll 2
    for (int k0 = 0; k0 < KD; k0 += 32) {
        // A 16x32 bf16 frag: elems 0..15 = K(k0+hs*16 ..) contiguous
        v8bf alo = *(const v8bf*)(Arow + k0 + hs * 16);
        v8bf ahi = *(const v8bf*)(Arow + k0 + hs * 16 + 8);
        v16bf a;
        #pragma unroll
        for (int e = 0; e < 8; ++e) { a[e] = alo[e]; a[e + 8] = ahi[e]; }
        // B 32x16 frags: lane n holds 16 contiguous K of W row (d*+n)  (W^T column)
        v16bf bi = *(const v16bf*)(Wi + k0);
        v16bf bf = *(const v16bf*)(Wf + k0);
        v16bf bg = *(const v16bf*)(Wg + k0);
        v16bf bo = *(const v16bf*)(Wo + k0);
        acc_i = __builtin_amdgcn_wmma_f32_16x16x32_bf16(false, a, false, bi, (short)0, acc_i, false, false);
        acc_f = __builtin_amdgcn_wmma_f32_16x16x32_bf16(false, a, false, bf, (short)0, acc_f, false, false);
        acc_g = __builtin_amdgcn_wmma_f32_16x16x32_bf16(false, a, false, bg, (short)0, acc_g, false, false);
        acc_o = __builtin_amdgcn_wmma_f32_16x16x32_bf16(false, a, false, bo, (short)0, acc_o, false, false);
    }

    // epilogue: C frag element v -> row b0 + v + hs*8, col d0 + colB
    const int dd = d0 + colB;
    const float bi_ = bias[dd], bf_ = bias[dd + 512], bg_ = bias[dd + 1024], bo_ = bias[dd + 1536];
    #pragma unroll
    for (int v = 0; v < 8; ++v) {
        int row = b0 + hs * 8 + v;
        float si = fast_sigmoid(acc_i[v] + bi_);
        float sf = fast_sigmoid(acc_f[v] + bf_);
        float tg = fast_tanh(acc_g[v] + bg_);
        float so = fast_sigmoid(acc_o[v] + bo_);
        size_t ix = (size_t)row * D + dd;
        float cn = sf * c[ix] + si * tg;
        float hn = so * fast_tanh(cn);
        c[ix] = cn;
        h[ix] = hn;
        Abf_out[(size_t)row * KD + dd] = f2bf(hn);   // h_bf for next step's GEMM
    }
}

// ---------------- attention: one workgroup per graph, single pass over x ----------------
// Flash-style online softmax per wave: each wave streams its nodes (full row held in
// registers, 16 floats/lane), maintains running (max, sum, weighted r) and rescales on a
// new max (wave-uniform branch). 8 per-wave partials merge via LDS. x is read ONCE.
__global__ __launch_bounds__(256) void attn_kernel(
    const float* __restrict__ x,        // [131072 x 512]
    const int*   __restrict__ seg,      // [513]
    const float* __restrict__ h,        // [512 x 512]  (q)
    u16*  __restrict__ Abf_out,         // write r_bf into cols [512,1024)
    float* __restrict__ out,            // q_star output on final step
    int write_out)
{
    __shared__ __align__(16) float q_s[D];
    __shared__ __align__(16) float r_part[8][D];   // 16 KB
    __shared__ float m_s[8], s_s[8];

    const int b = blockIdx.x;
    const int t = threadIdx.x;
    const int lane = t & 31, wave = t >> 5;

    const int s0 = seg[b];
    const int cnt = seg[b + 1] - s0;

    q_s[t]       = h[(size_t)b * D + t];
    q_s[t + 256] = h[(size_t)b * D + t + 256];
    __syncthreads();

    // lane-private q slice: lane l owns dims {k*128 + l*4 .. +4}
    v4f qv[4];
    #pragma unroll
    for (int k = 0; k < 4; ++k) qv[k] = *(const v4f*)(&q_s[k * 128 + lane * 4]);

    float m = -3.402823466e38f;
    float s = 0.f;
    v4f racc[4] = {};

    for (int i = wave; i < cnt; i += 8) {
        const float* __restrict__ xr = x + (size_t)(s0 + i) * D;
        v4f xv[4];
        float dot = 0.f;
        #pragma unroll
        for (int k = 0; k < 4; ++k) {
            xv[k] = *(const v4f*)(xr + k * 128 + lane * 4);
            dot += xv[k].x * qv[k].x + xv[k].y * qv[k].y
                 + xv[k].z * qv[k].z + xv[k].w * qv[k].w;
        }
        #pragma unroll
        for (int off = 16; off > 0; off >>= 1) dot += __shfl_xor(dot, off, 32);
        // dot is now identical on all lanes -> uniform rescale branch
        if (dot > m) {
            float sc = __expf(m - dot);
            s *= sc;
            #pragma unroll
            for (int k = 0; k < 4; ++k) racc[k] *= sc;
            m = dot;
        }
        float w = __expf(dot - m);
        s += w;
        #pragma unroll
        for (int k = 0; k < 4; ++k) racc[k] += w * xv[k];
    }

    // publish per-wave partials
    #pragma unroll
    for (int k = 0; k < 4; ++k) *(v4f*)(&r_part[wave][k * 128 + lane * 4]) = racc[k];
    if (lane == 0) { m_s[wave] = m; s_s[wave] = s; }
    __syncthreads();

    // merge 8 waves (flash merge): M = max m_w;  S = sum exp(m_w-M) s_w;  r = sum exp(m_w-M) r_w
    float M = -3.402823466e38f;
    #pragma unroll
    for (int w = 0; w < 8; ++w) M = fmaxf(M, m_s[w]);
    float S = 0.f, r0 = 0.f, r1 = 0.f;
    #pragma unroll
    for (int w = 0; w < 8; ++w) {
        float sc = __expf(m_s[w] - M);
        S  += sc * s_s[w];
        r0 += sc * r_part[w][t];
        r1 += sc * r_part[w][t + 256];
    }
    const float inv = 1.f / (S + 1e-16f);
    r0 *= inv; r1 *= inv;

    Abf_out[(size_t)b * KD + 512 + t]       = f2bf(r0);
    Abf_out[(size_t)b * KD + 512 + t + 256] = f2bf(r1);
    if (write_out) {
        out[(size_t)b * 1024 + t]             = q_s[t];
        out[(size_t)b * 1024 + t + 256]       = q_s[t + 256];
        out[(size_t)b * 1024 + 512 + t]       = r0;
        out[(size_t)b * 1024 + 512 + t + 256] = r1;
    }
}

extern "C" void kernel_launch(void* const* d_in, const int* in_sizes, int n_in,
                              void* d_out, int out_size, void* d_ws, size_t ws_size,
                              hipStream_t stream) {
    const float* x     = (const float*)d_in[0];
    const int*   batch = (const int*)d_in[1];
    const float* Wih   = (const float*)d_in[2];
    const float* Whh   = (const float*)d_in[3];
    const float* bih   = (const float*)d_in[4];
    const float* bhh   = (const float*)d_in[5];

    char* ws = (char*)d_ws;
    size_t off = 0;
    auto alloc = [&](size_t bytes) -> void* {
        void* p = ws + off;
        off += (bytes + 255) & ~(size_t)255;
        return p;
    };
    u16*   Wbf  = (u16*)  alloc((size_t)GD * KD * 2);   // 4 MB folded bf16 weights
    float* bias = (float*)alloc(GD * 4);
    u16*   Abf0 = (u16*)  alloc((size_t)NB * KD * 2);   // [h|r] bf16, double-buffered
    u16*   Abf1 = (u16*)  alloc((size_t)NB * KD * 2);
    float* c    = (float*)alloc((size_t)NB * D * 4);
    float* h    = (float*)alloc((size_t)NB * D * 4);
    int*   seg  = (int*)  alloc((NB + 1) * 4);
    u16*   Abf[2] = { Abf0, Abf1 };

    prep_weights<<<8192, 256, 0, stream>>>(Wih, Whh, bih, bhh, Wbf, bias);
    build_seg<<<3, 256, 0, stream>>>(batch, seg);
    init_state<<<2048, 256, 0, stream>>>(c, Abf0);

    for (int t = 0; t < T_STEPS; ++t) {
        lstm_kernel<<<128, 256, 0, stream>>>(Abf[t & 1], Wbf, bias, c, h, Abf[(t + 1) & 1]);
        attn_kernel<<<NB, 256, 0, stream>>>(x, seg, h, Abf[(t + 1) & 1],
                                            (float*)d_out, t == T_STEPS - 1 ? 1 : 0);
    }
}
